// Stage2_69982197121800
// MI455X (gfx1250) — matrix operand order repaired and checked by reading, not compiled, not run on hardware
//
#include <hip/hip_runtime.h>
#include <hip/hip_bf16.h>

typedef __attribute__((ext_vector_type(16))) __bf16 v16bf;
typedef __attribute__((ext_vector_type(8)))  __bf16 v8bf;
typedef __attribute__((ext_vector_type(8)))  float  v8f;
typedef __attribute__((ext_vector_type(4)))  float  v4f;
typedef __attribute__((ext_vector_type(2)))  float  v2f;

constexpr int Bn = 1024;      // batch
constexpr int Zn = 32768;     // features
constexpr int Dn = 128;       // dim
constexpr int BT = 16;        // batch rows per tile (WMMA M)
constexpr int NBT = Bn / BT;  // 64 batch tiles
constexpr int NSLICE = 8;     // z-slices per batch tile (workgroups)
constexpr int WAVES = 4;      // waves per workgroup
constexpr int NS = NSLICE * WAVES;            // 32 flash partials per batch tile
constexpr int ZPW = Zn / NS;                  // 1024 z per wave
constexpr int CHUNK = 32;                     // z per inner iteration (WMMA K of GEMM2)
constexpr int NCHUNK = ZPW / CHUNK;           // 32 chunks
constexpr int SPAD = 18;                      // padded float2 row stride (bank-conflict-free)

template <bool RA>
__device__ __forceinline__ v8f wmma_bf16(v16bf a, v16bf b, v8f c) {
  return __builtin_amdgcn_wmma_f32_16x16x32_bf16(false, a, false, b, (short)0, c, RA, false);
}

// ---------------- kernel 1: pack mask bits + per-row counts ----------------
__global__ void mask_pack_kernel(const int* __restrict__ zs,
                                 unsigned* __restrict__ maskw,
                                 float* __restrict__ counts) {
  const int b = blockIdx.x;
  const int tid = threadIdx.x;           // 256 threads = 8 waves
  const int lane = tid & 31, wid = tid >> 5;
  const int* row = zs + (size_t)b * Zn;
  int cnt = 0;
  for (int c = 0; c < Zn; c += 256) {
    const bool pred = row[c + tid] > 0;
    cnt += pred ? 1 : 0;
    const unsigned m = (unsigned)__ballot(pred);   // wave32: 32 z bits, word aligned
    if (lane == 0) maskw[(size_t)b * (Zn / 32) + (c >> 5) + wid] = m;
  }
  __shared__ int sred[256];
  sred[tid] = cnt;
  __syncthreads();
  for (int st = 128; st > 0; st >>= 1) {
    if (tid < st) sred[tid] += sred[tid + st];
    __syncthreads();
  }
  if (tid == 0) counts[b] = (float)sred[0];
}

// ------- kernel 2: fp32 -> bf16 into WMMA-fragment-native tiled layouts ----
// embBs: GEMM1 B operand. Block (zb, kk) of 32z x 32d:
//   idx = zb*4096 + kk*1024 + (z%32)*32 + (d%32)
// embTs: GEMM2 B operand. Block zb of 32z x 128d:
//   idx = zb*4096 + d*32 + (z%32)
__global__ void cvt_emb_kernel(const float* __restrict__ emb,
                               __bf16* __restrict__ embBs,
                               __bf16* __restrict__ embTs) {
  const size_t i = (size_t)blockIdx.x * 256 + threadIdx.x;   // Zn*Dn total
  const float v = emb[i];
  const __bf16 h = (__bf16)v;
  const size_t z = i >> 7, d = i & 127;
  const size_t zb = z >> 5, zi = z & 31;
  embBs[zb * 4096 + (d >> 5) * 1024 + zi * 32 + (d & 31)] = h;
  embTs[zb * 4096 + d * 32 + zi] = h;
}

__global__ void cvt_ctx_kernel(const float* __restrict__ ctx,
                               __bf16* __restrict__ ctxB) {
  const size_t i = (size_t)blockIdx.x * 256 + threadIdx.x;   // Bn*Dn total
  ctxB[i] = (__bf16)ctx[i];
}

// ---------------- kernel 3: fused flash (masked softmax attention) ---------
__global__ void __launch_bounds__(WAVES * 32)
flash_kernel(const __bf16* __restrict__ ctxB,
             const __bf16* __restrict__ embBs,
             const __bf16* __restrict__ embTs,
             const unsigned* __restrict__ maskw,
             float* __restrict__ pacc,
             float* __restrict__ pm,
             float* __restrict__ pl) {
  const float NEG_INF = -__builtin_inff();
  const float SCL = 1.4426950408889634f * 0.08838834764831845f; // log2(e)/sqrt(128)

  const int btile = blockIdx.x / NSLICE;
  const int slice = blockIdx.x % NSLICE;
  const int wid  = threadIdx.x >> 5;
  const int lane = threadIdx.x & 31;
  const int n16  = lane & 15;     // column within 16-wide tile / row for A-frags
  const int kh   = lane >> 4;     // half-wave id (K/M split per WMMA layouts)
  const int bbase = btile * BT;
  const int sidx  = slice * WAVES + wid;        // flash partial id [0,32)
  const int z0    = sidx * ZPW;

  __shared__ v2f  sbuf[WAVES][BT][SPAD];  // (s0,s1) pair transpose buffer (~9 KB)
  __shared__ float cbuf[WAVES][BT];       // per-row rescale broadcast

  // ---- load A fragments for ctx (kept in registers for whole kernel) ----
  v16bf afrag[4];
#pragma unroll
  for (int kk = 0; kk < 4; ++kk) {
    const __bf16* base = ctxB + (size_t)(bbase + n16) * Dn + 32 * kk + 16 * kh;
    v8bf lo = *(const v8bf*)base;
    v8bf hi = *(const v8bf*)(base + 8);
    afrag[kk] = __builtin_shufflevector(lo, hi, 0,1,2,3,4,5,6,7,8,9,10,11,12,13,14,15);
  }

  // ---- flash state: per-lane stats for row (lane&15) ----
  v8f acc[8];
#pragma unroll
  for (int t = 0; t < 8; ++t) acc[t] = (v8f){0.f,0.f,0.f,0.f,0.f,0.f,0.f,0.f};
  float m_ln = NEG_INF, l_ln = 0.f;

  for (int ch = 0; ch < NCHUNK; ++ch) {
    const int zbase = z0 + ch * CHUNK;
    const size_t blk = (size_t)(zbase >> 5) * 4096;   // 32z block offset in both tables

    // near prefetch: exactly the next 8 KB block of each table
    if (ch + 1 < NCHUNK) {
      __builtin_prefetch(embBs + blk + 4096 + lane * 128, 0, 3);
      __builtin_prefetch(embTs + blk + 4096 + lane * 128, 0, 3);
    }

    // ---- GEMM1: S[16x32] = ctx[16x128] x emb^T[128x32]  (two 16x16 C tiles) ----
    v8f s0 = (v8f){0.f,0.f,0.f,0.f,0.f,0.f,0.f,0.f};
    v8f s1 = (v8f){0.f,0.f,0.f,0.f,0.f,0.f,0.f,0.f};
#pragma unroll
    for (int kk = 0; kk < 4; ++kk) {
      const v16bf b0 = *(const v16bf*)(embBs + blk + kk * 1024 + n16 * 32 + 16 * kh);
      s0 = wmma_bf16<false>(afrag[kk], b0, s0);
      const v16bf b1 = *(const v16bf*)(embBs + blk + kk * 1024 + (16 + n16) * 32 + 16 * kh);
      s1 = wmma_bf16<false>(afrag[kk], b1, s1);
    }

    // ---- transpose raw scores through LDS (C layout -> per-lane row) ----
#pragma unroll
    for (int r = 0; r < 8; ++r)
      sbuf[wid][r + 8 * kh][n16] = (v2f){s0[r], s1[r]};   // one b64 per row
    asm volatile("s_wait_dscnt 0x0" ::: "memory");

    // q[i] = {se[2i], se[8+2i], se[2i+1], se[8+2i+1]}  (z, z+16 pairs)
    v4f q[4];
#pragma unroll
    for (int i = 0; i < 4; ++i)
      q[i] = *(const v4f*)&sbuf[wid][n16][8 * kh + 2 * i];

    // ---- mask + online softmax, all in-lane (row = n16) ----
    const unsigned w = maskw[(size_t)(bbase + n16) * (Zn >> 5) + (zbase >> 5)];
    float se[16];
#pragma unroll
    for (int i = 0; i < 4; ++i) {
      se[2 * i]         = q[i][0];
      se[8 + 2 * i]     = q[i][1];
      se[2 * i + 1]     = q[i][2];
      se[8 + 2 * i + 1] = q[i][3];
    }
#pragma unroll
    for (int e = 0; e < 16; ++e) {
      const int bit = (e < 8) ? (8 * kh + e) : (8 + 8 * kh + e);  // A-frag K position
      const float sc = se[e] * SCL;
      se[e] = ((w >> bit) & 1u) ? sc : NEG_INF;
    }
    float rmax = se[0];
#pragma unroll
    for (int e = 1; e < 16; ++e) rmax = fmaxf(rmax, se[e]);
    rmax = fmaxf(rmax, __shfl_xor(rmax, 16));        // combine the two K-halves
    const float nm = fmaxf(m_ln, rmax);
    const float corr = (m_ln == NEG_INF) ? 0.f : __builtin_amdgcn_exp2f(m_ln - nm);

    v16bf pf;
    float psum = 0.f;
#pragma unroll
    for (int e = 0; e < 16; ++e) {
      const float p = (se[e] == NEG_INF) ? 0.f : __builtin_amdgcn_exp2f(se[e] - nm);
      psum += p;
      pf[e] = (__bf16)p;                             // P directly in A-frag layout
    }
    psum += __shfl_xor(psum, 16);
    l_ln = l_ln * corr + psum;
    m_ln = nm;

    // ---- rescale accumulator only when some row saw a new max ----
    if (__ballot(corr != 1.0f)) {
      if (kh == 0) cbuf[wid][n16] = corr;
      asm volatile("s_wait_dscnt 0x0" ::: "memory");
      v4f c0 = *(const v4f*)&cbuf[wid][8 * kh];
      v4f c1 = *(const v4f*)&cbuf[wid][8 * kh + 4];
      float crr[8] = {c0[0], c0[1], c0[2], c0[3], c1[0], c1[1], c1[2], c1[3]};
#pragma unroll
      for (int t = 0; t < 8; ++t)
#pragma unroll
        for (int r = 0; r < 8; ++r) acc[t][r] *= crr[r];
    }

    // ---- GEMM2: acc[16x128] += P[16x32] x emb[32x128] (8 d-tiles) ----
    {
      const v16bf bt0 = __builtin_shufflevector(*(const v8bf*)(embTs + blk + n16 * 32 + 8 * kh), *(const v8bf*)(embTs + blk + n16 * 32 + 8 * kh + 16), 0,1,2,3,4,5,6,7,8,9,10,11,12,13,14,15);
      acc[0] = wmma_bf16<false>(pf, bt0, acc[0]);
#pragma unroll
      for (int t = 1; t < 8; ++t) {
        const v16bf bt = __builtin_shufflevector(*(const v8bf*)(embTs + blk + (16 * t + n16) * 32 + 8 * kh), *(const v8bf*)(embTs + blk + (16 * t + n16) * 32 + 8 * kh + 16), 0,1,2,3,4,5,6,7,8,9,10,11,12,13,14,15);
        acc[t] = wmma_bf16<true>(pf, bt, acc[t]);    // reuse_a: identical consecutive WMMA
      }
    }
  }

  // ---- write flash partials (per-lane contiguous 256B) ----
  float* pa = pacc + ((size_t)btile * NS + sidx) * BT * Dn + (size_t)lane * 64;
#pragma unroll
  for (int t = 0; t < 8; ++t) {
    v4f lo = {acc[t][0], acc[t][1], acc[t][2], acc[t][3]};
    v4f hi = {acc[t][4], acc[t][5], acc[t][6], acc[t][7]};
    *(v4f*)(pa + t * 8)     = lo;
    *(v4f*)(pa + t * 8 + 4) = hi;
  }
  if (kh == 0) {
    pm[((size_t)btile * NS + sidx) * BT + n16] = m_ln;
    pl[((size_t)btile * NS + sidx) * BT + n16] = l_ln;
  }
}

// ---------------- kernel 4: merge partials, apply 1/(l*count) --------------
// pacc element (slice s, row, d) lives at:
//   ((row>>3)*16 + (d&15))*64 + (d>>4)*8 + (row&7)   within the slice's 16x128 block
__global__ void reduce_kernel(const float* __restrict__ pacc,
                              const float* __restrict__ pm,
                              const float* __restrict__ pl,
                              const float* __restrict__ counts,
                              float* __restrict__ out) {
  const float NEG_INF = -__builtin_inff();
  const int btile = blockIdx.x;
  const int d  = threadIdx.x & 127;
  const int rh = threadIdx.x >> 7;       // 0..1
  const int frag = ((d & 15) << 6) + ((d >> 4) << 3);  // lane/reg part from d
  for (int i = 0; i < 8; ++i) {
    const int row = rh * 8 + i;
    const int b = btile * BT + row;
    const int idx = ((row >> 3) << 10) + frag + (row & 7);
    float M = NEG_INF;
    for (int s = 0; s < NS; ++s)
      M = fmaxf(M, pm[((size_t)btile * NS + s) * BT + row]);
    float L = 0.f, o = 0.f;
    for (int s = 0; s < NS; ++s) {
      const float ms = pm[((size_t)btile * NS + s) * BT + row];
      const float f = (ms == NEG_INF) ? 0.f : __builtin_amdgcn_exp2f(ms - M);
      L += f * pl[((size_t)btile * NS + s) * BT + row];
      o += f * pacc[((size_t)btile * NS + s) * BT * Dn + idx];
    }
    const float cnt = fmaxf(counts[b], 1.0f);
    out[(size_t)b * Dn + d] = (L > 0.f) ? o / (L * cnt) : 0.f;
  }
}

extern "C" void kernel_launch(void* const* d_in, const int* in_sizes, int n_in,
                              void* d_out, int out_size, void* d_ws, size_t ws_size,
                              hipStream_t stream) {
  (void)in_sizes; (void)n_in; (void)out_size; (void)ws_size;
  const int*   zs  = (const int*)d_in[0];
  const float* ctx = (const float*)d_in[1];
  const float* emb = (const float*)d_in[2];
  float* out = (float*)d_out;

  char* ws = (char*)d_ws;
  size_t off = 0;
  __bf16* embBs = (__bf16*)(ws + off); off += (size_t)Zn * Dn * 2;          // 8 MB
  __bf16* embTs = (__bf16*)(ws + off); off += (size_t)Zn * Dn * 2;          // 8 MB
  __bf16* ctxB  = (__bf16*)(ws + off); off += (size_t)Bn * Dn * 2;          // 256 KB
  unsigned* maskw = (unsigned*)(ws + off); off += (size_t)Bn * (Zn / 32) * 4; // 4 MB
  float* counts = (float*)(ws + off); off += (size_t)Bn * 4;
  float* pacc = (float*)(ws + off); off += (size_t)NBT * NS * BT * Dn * 4; // 16 MB
  float* pm = (float*)(ws + off); off += (size_t)NBT * NS * BT * 4;
  float* pl = (float*)(ws + off); off += (size_t)NBT * NS * BT * 4;

  cvt_emb_kernel<<<(Zn * Dn) / 256, 256, 0, stream>>>(emb, embBs, embTs);
  cvt_ctx_kernel<<<(Bn * Dn) / 256, 256, 0, stream>>>(ctx, ctxB);
  mask_pack_kernel<<<Bn, 256, 0, stream>>>(zs, maskw, counts);
  flash_kernel<<<NBT * NSLICE, WAVES * 32, 0, stream>>>(ctxB, embBs, embTs, maskw, pacc, pm, pl);
  reduce_kernel<<<NBT, 256, 0, stream>>>(pacc, pm, pl, counts, out);
}
